// StackedPVConv_85212151153325
// MI455X (gfx1250) — compile-verified, matrix operand order checked by reading, not run on hardware
//
#include <hip/hip_runtime.h>
#include <hip/hip_bf16.h>

// ---------------------------------------------------------------------------
// StackedPVConv for MI455X (gfx1250, wave32, WMMA)
//   conv3d 3x3x3 + pointwise linear as implicit GEMM via v_wmma_f32_16x16x32_f16.
//   Voxel grids staged channel-minor f16 with a zero halo ([B,(r+2)^3,Cpad]) so
//   the conv K-loop is branch-free: 6x global_load_b128 + 2x v_wmma per chunk.
// ---------------------------------------------------------------------------

typedef __attribute__((ext_vector_type(16))) _Float16 v16h;
typedef __attribute__((ext_vector_type(8)))  _Float16 v8h;
typedef __attribute__((ext_vector_type(8)))  float    v8f;

#define THREADS 256
#define GN_GROUPS 8
#define GN_EPS 1e-5f

#define CAT16(lo, hi) __builtin_shufflevector(lo, hi, 0, 1, 2, 3, 4, 5, 6, 7, 8, 9, 10, 11, 12, 13, 14, 15)

// ------------------------- utility kernels ---------------------------------

__global__ void zero_f32_kernel(float* __restrict__ p, long n) {
  long i = (long)blockIdx.x * blockDim.x + threadIdx.x;
  if (i < n) p[i] = 0.0f;
}

__global__ void zero_f16_kernel(_Float16* __restrict__ p, long n) {
  long i = (long)blockIdx.x * blockDim.x + threadIdx.x;
  if (i < n) p[i] = (_Float16)0.0f;
}

// conv weights f32 [oc][ic][27] -> f16 [oc][27*Cpad], K = tap*Cpad + c
__global__ void repack_conv_w_kernel(const float* __restrict__ src, _Float16* __restrict__ dst,
                                     int oc, int ic, int Cpad) {
  long i = (long)blockIdx.x * blockDim.x + threadIdx.x;
  int Ktot = 27 * Cpad;
  if (i >= (long)oc * Ktot) return;
  int m = (int)(i / Ktot), k = (int)(i % Ktot);
  int t = k / Cpad, c = k % Cpad;
  float v = (c < ic) ? src[((long)m * ic + c) * 27 + t] : 0.0f;
  dst[i] = (_Float16)v;
}

// pointwise weights f32 [oc][ic] -> f16 [oc][Kpad]
__global__ void repack_pw_kernel(const float* __restrict__ src, _Float16* __restrict__ dst,
                                 int oc, int ic, int Kpad) {
  long i = (long)blockIdx.x * blockDim.x + threadIdx.x;
  if (i >= (long)oc * Kpad) return;
  int m = (int)(i / Kpad), k = (int)(i % Kpad);
  dst[i] = (_Float16)((k < ic) ? src[(long)m * ic + k] : 0.0f);
}

// coords [B,N,3] -> featF f32 [B,3,N], featH f16 [B,N,Cpad] (zero-padded)
__global__ void init_feat_kernel(const float* __restrict__ coords,
                                 float* __restrict__ f32, _Float16* __restrict__ f16,
                                 int B, int N, int Cpad) {
  long i = (long)blockIdx.x * blockDim.x + threadIdx.x;
  if (i >= (long)B * N * Cpad) return;
  int c = (int)(i % Cpad);
  long bn = i / Cpad;
  int n = (int)(bn % N), b = (int)(bn / N);
  float v = (c < 3) ? coords[((long)b * N + n) * 3 + c] : 0.0f;
  f16[i] = (_Float16)v;
  if (c < 3) f32[((long)b * 3 + c) * N + n] = v;
}

// Per-batch: mean-center, scale by 2*max||c||, +0.5 -> cbase [B,3,N]
__global__ void coords_norm_kernel(const float* __restrict__ coords,
                                   float* __restrict__ cbase, int N) {
  int b = blockIdx.x, tid = threadIdx.x;
  __shared__ float sA[THREADS], sB[THREADS], sC[THREADS];
  __shared__ float mean3[3], maxn_s;
  float sx = 0, sy = 0, sz = 0;
  for (int n = tid; n < N; n += THREADS) {
    const float* p = coords + ((long)b * N + n) * 3;
    sx += p[0]; sy += p[1]; sz += p[2];
  }
  sA[tid] = sx; sB[tid] = sy; sC[tid] = sz;
  __syncthreads();
  for (int st = THREADS / 2; st > 0; st >>= 1) {
    if (tid < st) { sA[tid] += sA[tid + st]; sB[tid] += sB[tid + st]; sC[tid] += sC[tid + st]; }
    __syncthreads();
  }
  if (tid == 0) { mean3[0] = sA[0] / N; mean3[1] = sB[0] / N; mean3[2] = sC[0] / N; }
  __syncthreads();
  float mx = 0.0f;
  for (int n = tid; n < N; n += THREADS) {
    const float* p = coords + ((long)b * N + n) * 3;
    float dx = p[0] - mean3[0], dy = p[1] - mean3[1], dz = p[2] - mean3[2];
    mx = fmaxf(mx, sqrtf(dx * dx + dy * dy + dz * dz));
  }
  sA[tid] = mx;
  __syncthreads();
  for (int st = THREADS / 2; st > 0; st >>= 1) {
    if (tid < st) sA[tid] = fmaxf(sA[tid], sA[tid + st]);
    __syncthreads();
  }
  if (tid == 0) maxn_s = sA[0];
  __syncthreads();
  float inv = 1.0f / (2.0f * maxn_s);
  for (int n = tid; n < N; n += THREADS) {
    const float* p = coords + ((long)b * N + n) * 3;
    cbase[((long)b * 3 + 0) * N + n] = (p[0] - mean3[0]) * inv + 0.5f;
    cbase[((long)b * 3 + 1) * N + n] = (p[1] - mean3[1]) * inv + 0.5f;
    cbase[((long)b * 3 + 2) * N + n] = (p[2] - mean3[2]) * inv + 0.5f;
  }
}

// ------------------------- voxelize ----------------------------------------

__device__ __forceinline__ int vox_coord(float cb, int r) {
  float nc = fminf(fmaxf(cb * (float)r, 0.0f), (float)(r - 1));
  return (int)rintf(nc);
}

__global__ void scatter_kernel(const float* __restrict__ feat,   // [B,C,N]
                               const float* __restrict__ cbase,  // [B,3,N]
                               float* __restrict__ sum,          // [B,C,r^3]
                               float* __restrict__ cnt,          // [B,r^3]
                               int B, int C, int N, int r) {
  long i = (long)blockIdx.x * blockDim.x + threadIdx.x;
  if (i >= (long)B * N) return;
  int b = (int)(i / N), n = (int)(i % N);
  int vx = vox_coord(cbase[((long)b * 3 + 0) * N + n], r);
  int vy = vox_coord(cbase[((long)b * 3 + 1) * N + n], r);
  int vz = vox_coord(cbase[((long)b * 3 + 2) * N + n], r);
  int R3 = r * r * r;
  int idx = (vx * r + vy) * r + vz;
  atomicAdd(&cnt[(long)b * R3 + idx], 1.0f);
  for (int c = 0; c < C; ++c)
    atomicAdd(&sum[((long)b * C + c) * R3 + idx], feat[((long)b * C + c) * N + n]);
}

// sum [B,C,r^3] / cnt -> g16 [B,(r+2)^3,Cpad] channel-minor with zero halo
__global__ void vox_finalize_kernel(const float* __restrict__ sum,
                                    const float* __restrict__ cnt,
                                    _Float16* __restrict__ g16,
                                    int B, int C, int Cpad, int r) {
  int rp = r + 2;
  long R3p = (long)rp * rp * rp, R3 = (long)r * r * r;
  long i = (long)blockIdx.x * blockDim.x + threadIdx.x;
  if (i >= (long)B * R3p * Cpad) return;
  int c = (int)(i % Cpad);
  long bv = i / Cpad;
  int pv = (int)(bv % R3p), b = (int)(bv / R3p);
  int z = pv % rp, y = (pv / rp) % rp, x = pv / (rp * rp);
  float val = 0.0f;
  if (c < C && x >= 1 && x <= r && y >= 1 && y <= r && z >= 1 && z <= r) {
    int v = (((x - 1) * r + (y - 1)) * r + (z - 1));
    val = sum[((long)b * C + c) * R3 + v] / fmaxf(cnt[(long)b * R3 + v], 1.0f);
  }
  g16[i] = (_Float16)val;
}

// ------------------------- WMMA conv3d (implicit GEMM) ---------------------
// A: weights f16 [Cout][27*CPAD]; B: halo grid f16 [B][(r+2)^3][CPAD].
// Two N-tiles (voxels) per wave, A fragment reused. Branch-free K-loop.
// out f32 [B][Cout][r^3].  Cout %16==0, r^3 %32==0.

template <int CPAD>
__global__ void __launch_bounds__(32)
conv3d_wmma_kernel(const _Float16* __restrict__ w,
                   const _Float16* __restrict__ g,
                   const float* __restrict__ bias,
                   float* __restrict__ out,
                   int Cout, int r) {
  const int R3 = r * r * r;
  const int rp = r + 2;
  const long R3p = (long)rp * rp * rp;
  const int ntiles = R3 >> 5;
  const int mtiles = Cout >> 4;
  int bid = blockIdx.x;
  int nt = bid % ntiles; bid /= ntiles;
  int mt = bid % mtiles;
  int b = bid / mtiles;
  int lane = threadIdx.x & 31;
  int l15 = lane & 15, hi = lane >> 4;
  int ng0 = nt * 32 + l15, ng1 = ng0 + 16;
  int vx0 = ng0 / (r * r), vy0 = (ng0 / r) % r, vz0 = ng0 % r;
  int vx1 = ng1 / (r * r), vy1 = (ng1 / r) % r, vz1 = ng1 % r;
  long pv0 = ((long)(vx0 + 1) * rp + (vy0 + 1)) * rp + (vz0 + 1);
  long pv1 = ((long)(vx1 + 1) * rp + (vy1 + 1)) * rp + (vz1 + 1);
  const _Float16* wrow = w + (long)(mt * 16 + l15) * (27 * CPAD) + 8 * hi;
  const _Float16* gq = g + (long)b * R3p * CPAD + 8 * hi;
  v8f acc0 = {}, acc1 = {};
  int k = 0;
  for (int dx = -1; dx <= 1; ++dx)
    for (int dy = -1; dy <= 1; ++dy)
      for (int dz = -1; dz <= 1; ++dz) {
        long doff = ((long)dx * rp + dy) * rp + dz;
        long o0 = (pv0 + doff) * CPAD;
        long o1 = (pv1 + doff) * CPAD;
#pragma unroll
        for (int ic0 = 0; ic0 < CPAD; ic0 += 32, k += 32) {
          v8h a0 = *(const v8h*)(wrow + k);
          v8h a1 = *(const v8h*)(wrow + k + 16);
          v8h b00 = *(const v8h*)(gq + o0 + ic0);
          v8h b01 = *(const v8h*)(gq + o0 + ic0 + 16);
          v8h b10 = *(const v8h*)(gq + o1 + ic0);
          v8h b11 = *(const v8h*)(gq + o1 + ic0 + 16);
          v16h af = CAT16(a0, a1);
          acc0 = __builtin_amdgcn_wmma_f32_16x16x32_f16(false, af, false, CAT16(b00, b01),
                                                        (short)0, acc0, false, false);
          acc1 = __builtin_amdgcn_wmma_f32_16x16x32_f16(false, af, false, CAT16(b10, b11),
                                                        (short)0, acc1, false, false);
        }
      }
  int mb = hi * 8;
#pragma unroll
  for (int j = 0; j < 8; ++j) {
    int m = mt * 16 + mb + j;
    float bi = bias[m];
    out[((long)b * Cout + m) * R3 + ng0] = acc0[j] + bi;
    out[((long)b * Cout + m) * R3 + ng1] = acc1[j] + bi;
  }
}

// ------------------------- WMMA pointwise GEMM -----------------------------
// A: pw f16 [Cout][Cpad]; B: features f16 [B][N][Cpad]; out f32 [B][Cout][N].
// Cpad %32==0 (zero-padded), so the K-loop is guard-free.

__global__ void __launch_bounds__(32)
pointwise_wmma_kernel(const _Float16* __restrict__ w,
                      const _Float16* __restrict__ f,
                      const float* __restrict__ bias,
                      float* __restrict__ out,
                      int Cpad, int Cout, int N) {
  const int ntiles = N >> 4;
  const int mtiles = Cout >> 4;
  int bid = blockIdx.x;
  int nt = bid % ntiles; bid /= ntiles;
  int mt = bid % mtiles;
  int b = bid / mtiles;
  int lane = threadIdx.x & 31;
  int l15 = lane & 15, hi = lane >> 4;
  int ng = nt * 16 + l15;
  const _Float16* wrow = w + (long)(mt * 16 + l15) * Cpad + 8 * hi;
  const _Float16* frow = f + ((long)b * N + ng) * Cpad + 8 * hi;
  v8f acc = {};
  for (int k0 = 0; k0 < Cpad; k0 += 32) {
    v8h a0 = *(const v8h*)(wrow + k0);
    v8h a1 = *(const v8h*)(wrow + k0 + 16);
    v8h b0 = *(const v8h*)(frow + k0);
    v8h b1 = *(const v8h*)(frow + k0 + 16);
    acc = __builtin_amdgcn_wmma_f32_16x16x32_f16(false, CAT16(a0, a1), false, CAT16(b0, b1),
                                                 (short)0, acc, false, false);
  }
  int mb = hi * 8;
#pragma unroll
  for (int j = 0; j < 8; ++j) {
    int m = mt * 16 + mb + j;
    out[((long)b * Cout + m) * N + ng] = acc[j] + bias[m];
  }
}

// ------------------------- groupnorm (+ optional swish) --------------------
// x [B,C,S] in-place; optional f16 copy into halo layout [B,(r+2)^3,Cpad16].

__global__ void groupnorm_kernel(float* __restrict__ x, _Float16* __restrict__ y16,
                                 const float* __restrict__ gm, const float* __restrict__ bt,
                                 int C, long S, int do_swish, int Cpad16, int r) {
  int bg = blockIdx.x;
  int gr = bg % GN_GROUPS, b = bg / GN_GROUPS;
  int cpg = C / GN_GROUPS;
  long n = (long)cpg * S;
  float* base = x + ((long)b * C + (long)gr * cpg) * S;
  __shared__ float s1[THREADS], s2[THREADS];
  float sum = 0, sq = 0;
  for (long i = threadIdx.x; i < n; i += THREADS) {
    float v = base[i];
    sum += v; sq += v * v;
  }
  s1[threadIdx.x] = sum; s2[threadIdx.x] = sq;
  __syncthreads();
  for (int st = THREADS / 2; st > 0; st >>= 1) {
    if ((int)threadIdx.x < st) {
      s1[threadIdx.x] += s1[threadIdx.x + st];
      s2[threadIdx.x] += s2[threadIdx.x + st];
    }
    __syncthreads();
  }
  float mean = s1[0] / (float)n;
  float var = s2[0] / (float)n - mean * mean;
  float rstd = rsqrtf(var + GN_EPS);
  int rp = r + 2;
  long R3p = (long)rp * rp * rp;
  for (long i = threadIdx.x; i < n; i += THREADS) {
    int c = gr * cpg + (int)(i / S);
    float v = (base[i] - mean) * rstd * gm[c] + bt[c];
    if (do_swish) v = v / (1.0f + __expf(-v));
    base[i] = v;
    if (y16) {
      int s = (int)(i % S);
      int z = s % r, y = (s / r) % r, xg = s / (r * r);
      long pv = ((long)(xg + 1) * rp + (y + 1)) * rp + (z + 1);
      y16[((long)b * R3p + pv) * Cpad16 + c] = (_Float16)v;
    }
  }
}

// ------------------------- trilinear devoxelize ----------------------------

__global__ void devox_kernel(const float* __restrict__ grid,   // [B,C,r^3]
                             const float* __restrict__ cbase,  // [B,3,N]
                             float* __restrict__ out,          // [B,C,N]
                             int B, int C, int N, int r) {
  long i = (long)blockIdx.x * blockDim.x + threadIdx.x;
  if (i >= (long)B * C * N) return;
  int n = (int)(i % N);
  int c = (int)((i / N) % C);
  int b = (int)(i / ((long)C * N));
  int R3 = r * r * r;
  float fc[3]; int c0[3], c1[3]; float d[3];
#pragma unroll
  for (int k = 0; k < 3; ++k) {
    float v = cbase[((long)b * 3 + k) * N + n] * (float)r;
    fc[k] = fminf(fmaxf(v, 0.0f), (float)(r - 1));
    int f0 = (int)floorf(fc[k]);
    f0 = max(0, min(f0, r - 1));
    c0[k] = f0;
    c1[k] = min(f0 + 1, r - 1);
    d[k] = fc[k] - (float)f0;
  }
  const float* g = grid + ((long)b * C + c) * R3;
  float acc = 0.0f;
#pragma unroll
  for (int q = 0; q < 8; ++q) {
    int ix = (q & 4) ? c1[0] : c0[0];
    int iy = (q & 2) ? c1[1] : c0[1];
    int iz = (q & 1) ? c1[2] : c0[2];
    float wx = (q & 4) ? d[0] : 1.0f - d[0];
    float wy = (q & 2) ? d[1] : 1.0f - d[1];
    float wz = (q & 1) ? d[2] : 1.0f - d[2];
    acc += g[(ix * r + iy) * r + iz] * (wx * wy * wz);
  }
  out[i] = acc;
}

// ------------------------- epilogues ---------------------------------------

// featF [B,C,N] f32 = vf+pt ; featH [B,N,C] f16 (C==64 exactly)
__global__ void add_feat_kernel(const float* __restrict__ a, const float* __restrict__ bsrc,
                                float* __restrict__ of, _Float16* __restrict__ oh,
                                int B, int C, int N) {
  long i = (long)blockIdx.x * blockDim.x + threadIdx.x;
  if (i >= (long)B * C * N) return;
  int n = (int)(i % N);
  int c = (int)((i / N) % C);
  int b = (int)(i / ((long)C * N));
  float v = a[i] + bsrc[i];
  of[i] = v;
  oh[((long)b * N + n) * C + c] = (_Float16)v;
}

// out[b,n,c] = vfeat[b,c,n] + pt[b,c,n]
__global__ void add_transpose_kernel(const float* __restrict__ vf, const float* __restrict__ pt,
                                     float* __restrict__ out, int B, int C, int N) {
  long i = (long)blockIdx.x * blockDim.x + threadIdx.x;
  if (i >= (long)B * C * N) return;
  int n = (int)(i % N);
  int c = (int)((i / N) % C);
  int b = (int)(i / ((long)C * N));
  out[((long)b * N + n) * C + c] = vf[i] + pt[i];
}

// ---------------------------------------------------------------------------

static inline long cdiv(long a, long b) { return (a + b - 1) / b; }

static void launch_conv(int CPAD, const _Float16* w, const _Float16* g, const float* bias,
                        float* out, int B, int Cout, int r, hipStream_t stream) {
  long blocks = (long)B * (Cout >> 4) * ((r * r * r) >> 5);
  if (CPAD == 32)
    conv3d_wmma_kernel<32><<<blocks, 32, 0, stream>>>(w, g, bias, out, Cout, r);
  else if (CPAD == 64)
    conv3d_wmma_kernel<64><<<blocks, 32, 0, stream>>>(w, g, bias, out, Cout, r);
  else
    conv3d_wmma_kernel<128><<<blocks, 32, 0, stream>>>(w, g, bias, out, Cout, r);
}

extern "C" void kernel_launch(void* const* d_in, const int* in_sizes, int n_in,
                              void* d_out, int out_size, void* d_ws, size_t ws_size,
                              hipStream_t stream) {
  (void)in_sizes; (void)n_in; (void)out_size; (void)ws_size;
  const int B = 8, N = 4096;
  const int RES[3] = {32, 16, 8};
  const int CH[4]  = {3, 64, 64, 128};

  const float* coords = (const float*)d_in[0];

  char* ws = (char*)d_ws;
  auto alloc = [&](size_t bytes) -> char* {
    char* p = ws;
    ws += (bytes + 255) & ~(size_t)255;
    return p;
  };
  const long R3MAX = 32 * 32 * 32;
  const long R3PMAX = 34L * 34 * 34;
  float*    cbase = (float*)alloc((size_t)B * 3 * N * 4);
  float*    featF = (float*)alloc((size_t)B * 64 * N * 4);          // [B,C,N]
  _Float16* featH = (_Float16*)alloc((size_t)B * N * 64 * 2);       // [B,N,Cpad]
  float*    gA    = (float*)alloc((size_t)B * 64 * R3MAX * 4);      // conv1 out / gn1 [B,C,r^3]
  float*    gB    = (float*)alloc((size_t)B * 64 * R3MAX * 4);      // scatter sum / conv2 out / gn2
  _Float16* g16   = (_Float16*)alloc((size_t)B * R3PMAX * 64 * 2);  // gn1 f16 halo [B,(r+2)^3,C]
  _Float16* gIn16 = (_Float16*)alloc((size_t)B * R3PMAX * 32 * 2);  // voxelized f16 halo
  float*    cnt   = (float*)alloc((size_t)B * R3MAX * 4);
  float*    vfeat = (float*)alloc((size_t)B * 128 * N * 4);
  float*    ptb   = (float*)alloc((size_t)B * 128 * N * 4);
  _Float16* w16a  = (_Float16*)alloc((size_t)128 * 27 * 64 * 2);    // conv1 w [oc][27*Cpad]
  _Float16* w16b  = (_Float16*)alloc((size_t)128 * 27 * 128 * 2);   // conv2 w [oc][27*Cpad]
  _Float16* pw16  = (_Float16*)alloc((size_t)128 * 64 * 2);         // point w [oc][Cpad]

  // init: features = coords^T ; normalized base coords
  init_feat_kernel<<<cdiv((long)B * N * 32, THREADS), THREADS, 0, stream>>>(coords, featF, featH, B, N, 32);
  coords_norm_kernel<<<B, THREADS, 0, stream>>>(coords, cbase, N);

  for (int li = 0; li < 3; ++li) {
    const int ic = CH[li], oc = CH[li + 1], r = RES[li];
    const long R3 = (long)r * r * r;
    const int rp = r + 2;
    const long R3p = (long)rp * rp * rp;
    const int base = 1 + li * 12;
    const float* vw1 = (const float*)d_in[base + 0];
    const float* vb1 = (const float*)d_in[base + 1];
    const float* g1w = (const float*)d_in[base + 2];
    const float* g1b = (const float*)d_in[base + 3];
    const float* vw2 = (const float*)d_in[base + 4];
    const float* vb2 = (const float*)d_in[base + 5];
    const float* g2w = (const float*)d_in[base + 6];
    const float* g2b = (const float*)d_in[base + 7];
    const float* pw  = (const float*)d_in[base + 8];
    const float* pb  = (const float*)d_in[base + 9];
    const float* pgw = (const float*)d_in[base + 10];
    const float* pgb = (const float*)d_in[base + 11];

    const int CPAD1 = (li == 0) ? 32 : 64;   // conv1 input channels (padded)
    const int CPAD2 = oc;                    // conv2 input channels (64/128)
    const int CPADP = (li == 0) ? 32 : 64;   // point input channels (padded)

    // ---- voxelize: scatter-average into gB, finalize to f16 halo grid
    zero_f32_kernel<<<cdiv((long)B * ic * R3, THREADS), THREADS, 0, stream>>>(gB, (long)B * ic * R3);
    zero_f32_kernel<<<cdiv((long)B * R3, THREADS), THREADS, 0, stream>>>(cnt, (long)B * R3);
    scatter_kernel<<<cdiv((long)B * N, THREADS), THREADS, 0, stream>>>(featF, cbase, gB, cnt, B, ic, N, r);
    vox_finalize_kernel<<<cdiv((long)B * R3p * CPAD1, THREADS), THREADS, 0, stream>>>(
        gB, cnt, gIn16, B, ic, CPAD1, r);

    // ---- conv1 (WMMA) + gn + swish (gn emits f16 halo grid for conv2)
    repack_conv_w_kernel<<<cdiv((long)oc * 27 * CPAD1, THREADS), THREADS, 0, stream>>>(
        vw1, w16a, oc, ic, CPAD1);
    launch_conv(CPAD1, w16a, gIn16, vb1, gA, B, oc, r, stream);
    zero_f16_kernel<<<cdiv((long)B * R3p * CPAD2, THREADS), THREADS, 0, stream>>>(g16, (long)B * R3p * CPAD2);
    groupnorm_kernel<<<B * GN_GROUPS, THREADS, 0, stream>>>(gA, g16, g1w, g1b, oc, R3, 1, CPAD2, r);

    // ---- conv2 (WMMA) + gn
    repack_conv_w_kernel<<<cdiv((long)oc * 27 * CPAD2, THREADS), THREADS, 0, stream>>>(
        vw2, w16b, oc, oc, CPAD2);
    launch_conv(CPAD2, w16b, g16, vb2, gB, B, oc, r, stream);
    groupnorm_kernel<<<B * GN_GROUPS, THREADS, 0, stream>>>(gB, (_Float16*)nullptr, g2w, g2b, oc, R3, 0, 0, r);

    // ---- trilinear devoxelize -> vfeat
    devox_kernel<<<cdiv((long)B * oc * N, THREADS), THREADS, 0, stream>>>(gB, cbase, vfeat, B, oc, N, r);

    // ---- point branch: WMMA GEMM + gn + swish
    repack_pw_kernel<<<cdiv((long)oc * CPADP, THREADS), THREADS, 0, stream>>>(pw, pw16, oc, ic, CPADP);
    pointwise_wmma_kernel<<<(long)B * (oc >> 4) * (N >> 4), 32, 0, stream>>>(
        pw16, featH, pb, ptb, CPADP, oc, N);
    groupnorm_kernel<<<B * GN_GROUPS, THREADS, 0, stream>>>(ptb, (_Float16*)nullptr, pgw, pgb, oc, (long)N, 1, 0, r);

    // ---- combine
    if (li < 2) {
      add_feat_kernel<<<cdiv((long)B * oc * N, THREADS), THREADS, 0, stream>>>(
          vfeat, ptb, featF, featH, B, oc, N);
    } else {
      add_transpose_kernel<<<cdiv((long)B * oc * N, THREADS), THREADS, 0, stream>>>(
          vfeat, ptb, (float*)d_out, B, oc, N);
    }
  }
}
